// Beam_Search_79894981640635
// MI455X (gfx1250) — compile-verified, matrix operand order checked by reading, not run on hardware
//
#include <hip/hip_runtime.h>
#include <hip/hip_bf16.h>

// ---------------------------------------------------------------------------
// Beam search decoder (GRU cell + vocab projection + top-K=3 beam update)
// bf16 WMMA (v_wmma_f32_16x16x32_bf16) for all GEMMs, f32 accumulation.
// Async LDS staging (global_load_async_to_lds_b128 + s_wait_asynccnt) for the
// activation panel of the vocab projection.
// B=256, K=3, H=1024, E=512, V=32000, T=12 steps.
// ---------------------------------------------------------------------------

#define BATCH 256
#define KBEAM 3
#define NROW  (BATCH * KBEAM)   // 768
#define HID   1024
#define EMB   512
#define XIN   (EMB + HID)       // 1536
#define G3H   (3 * HID)         // 3072
#define VOC   32000
#define TLEN  12
#define SOS_T 1
#define EOS_T 2

#define NSPLIT 25               // V split into 25 chunks of 20 groups (64 cols each)
#define GROUPS_PER_SPLIT 20     // 500 groups total / 25

typedef __attribute__((ext_vector_type(16))) __bf16 v16bf;
typedef __attribute__((ext_vector_type(8)))  float  v8f;

// ---------------------------------------------------------------------------
// WMMA fragment load: 16x32 bf16 tile, row-major with leading dim ld.
// CDNA5 16-bit A layout (wave32): lane l(0..15)->row l, K base 0; lane 16..31
// -> row l-16, K base 8. VGPR v holds K pair: v<4 : kb+2v ; v>=4 : kb+16+2(v-4)
// B fragments mirror (lane->column), so the transposed weight uses same math.
// ---------------------------------------------------------------------------
__device__ __forceinline__ v16bf ld_frag(const __bf16* p, int ld) {
    const int l  = threadIdx.x & 31;
    const int r  = l & 15;
    const int kb = (l >> 4) * 8;
    v16bf f;
#pragma unroll
    for (int i = 0; i < 8; ++i) {
        const int kk = (i < 4) ? (kb + 2 * i) : (kb + 8 + 2 * i);
        f[2 * i]     = p[r * ld + kk];
        f[2 * i + 1] = p[r * ld + kk + 1];
    }
    return f;
}

__device__ __forceinline__ v8f wmma_bf16(v16bf a, v16bf b, v8f c) {
    return __builtin_amdgcn_wmma_f32_16x16x32_bf16(false, a, false, b,
                                                   (short)0, c, false, false);
}

// ---------------------------------------------------------------------------
// Weight convert + transpose:  BT[n*Kd + k] = bf16( W[k*N + n] )
// ---------------------------------------------------------------------------
__global__ void k_cvtT(const float* __restrict__ W, __bf16* __restrict__ BT,
                       int Kd, int N) {
    long idx = (long)blockIdx.x * blockDim.x + threadIdx.x;
    if (idx >= (long)Kd * N) return;
    int n = (int)(idx / Kd);
    int k = (int)(idx % Kd);
    BT[idx] = (__bf16)W[(long)k * N + n];
}

// ---------------------------------------------------------------------------
// Init beam state + hidden = broadcast(enc) + enc half of A_x (static part)
// ---------------------------------------------------------------------------
__global__ void k_init(const float* __restrict__ enc, float* __restrict__ hidden,
                       __bf16* __restrict__ Ax, int* __restrict__ tokens,
                       float* __restrict__ cum, int* __restrict__ fin,
                       int* __restrict__ seqs) {
    int r = blockIdx.x;            // 0..767
    int b = r / KBEAM, k = r % KBEAM;
    for (int i = threadIdx.x; i < HID; i += blockDim.x) {
        float v = enc[b * HID + i];
        hidden[r * HID + i] = v;
        Ax[r * XIN + EMB + i] = (__bf16)v;
    }
    if (threadIdx.x == 0) {
        tokens[r] = SOS_T;
        cum[r]    = (k == 0) ? 1.0f : 0.0f;
        fin[r]    = 0;
    }
    for (int i = threadIdx.x; i < TLEN; i += blockDim.x) seqs[r * TLEN + i] = EOS_T;
}

// ---------------------------------------------------------------------------
// Per-step gather: embed rows for current tokens (-> bf16 A_x head) and
// hidden f32 -> bf16 A for the Wh GEMM.
// ---------------------------------------------------------------------------
__global__ void k_gather_x(const float* __restrict__ embed,
                           const float* __restrict__ hidden,
                           const int* __restrict__ tokens,
                           __bf16* __restrict__ Ax, __bf16* __restrict__ Hb) {
    int r = blockIdx.x;
    int tok = tokens[r];
    const float* er = embed + (long)tok * EMB;
    for (int i = threadIdx.x; i < EMB; i += blockDim.x)
        Ax[r * XIN + i] = (__bf16)er[i];
    for (int i = threadIdx.x; i < HID; i += blockDim.x)
        Hb[r * HID + i] = (__bf16)hidden[r * HID + i];
}

// ---------------------------------------------------------------------------
// Register-blocked WMMA GEMM: one wave computes a 64x64 output tile
// (4x4 grid of 16x16 WMMA tiles -> 16 v_wmma per K-step, 1 b128 load each).
// C[M,N] = A[M,Kd](bf16,row-major) x BT[N,Kd](bf16) + bias
// ---------------------------------------------------------------------------
__global__ __launch_bounds__(32)
void k_gemm64(const __bf16* __restrict__ A, const __bf16* __restrict__ BT,
              const float* __restrict__ bias, float* __restrict__ C,
              int N, int Kd) {
    const int tn = blockIdx.x;   // 64-col group
    const int tm = blockIdx.y;   // 64-row group
    const __bf16* Ab = A  + (long)tm * 64 * Kd;
    const __bf16* Bb = BT + (long)tn * 64 * Kd;
    v8f acc[4][4] = {};
    for (int k = 0; k < Kd; k += 32) {
        v16bf a[4], b[4];
#pragma unroll
        for (int i = 0; i < 4; ++i) a[i] = ld_frag(Ab + (long)i * 16 * Kd + k, Kd);
#pragma unroll
        for (int j = 0; j < 4; ++j) b[j] = ld_frag(Bb + (long)j * 16 * Kd + k, Kd);
#pragma unroll
        for (int i = 0; i < 4; ++i)
#pragma unroll
            for (int j = 0; j < 4; ++j)
                acc[i][j] = wmma_bf16(a[i], b[j], acc[i][j]);
    }
    const int l   = threadIdx.x & 31;
    const int col = l & 15;
    const int rb  = (l >> 4) * 8;
#pragma unroll
    for (int j = 0; j < 4; ++j) {
        const float bv = bias[tn * 64 + j * 16 + col];
#pragma unroll
        for (int i = 0; i < 4; ++i) {
#pragma unroll
            for (int e = 0; e < 8; ++e)
                C[(long)(tm * 64 + i * 16 + rb + e) * N + tn * 64 + j * 16 + col] =
                    acc[i][j][e] + bv;
        }
    }
}

// ---------------------------------------------------------------------------
// GRU elementwise combine (PyTorch gate order r,z,n).
// ---------------------------------------------------------------------------
__global__ void k_gru(const float* __restrict__ gx, const float* __restrict__ gh,
                      const float* __restrict__ hidden, float* __restrict__ hnew,
                      __bf16* __restrict__ hnewb) {
    int idx = blockIdx.x * blockDim.x + threadIdx.x;
    if (idx >= NROW * HID) return;
    int r = idx >> 10, c = idx & (HID - 1);
    const float* gxr = gx + (long)r * G3H;
    const float* ghr = gh + (long)r * G3H;
    float rg = 1.0f / (1.0f + __expf(-(gxr[c] + ghr[c])));
    float zg = 1.0f / (1.0f + __expf(-(gxr[HID + c] + ghr[HID + c])));
    float n  = tanhf(gxr[2 * HID + c] + rg * ghr[2 * HID + c]);
    float h  = hidden[idx];
    float o  = (1.0f - zg) * n + zg * h;
    hnew[idx]  = o;
    hnewb[idx] = (__bf16)o;
}

// ---------------------------------------------------------------------------
// Vocab projection GEMM fused with online softmax stats + per-row top-3.
// grid = (48 m-tiles, NSPLIT). One wave per block.
//  - A panel (16x1024 bf16 = 32KB) is staged to LDS ONCE via CDNA5 async
//    copy (global_load_async_to_lds_b128, fenced by s_wait_asynccnt) and
//    re-read for all 20 column groups.
//  - Each group computes 16x64 logits (1 A-frag from LDS reused over 4
//    B-frags), then lanes 0..15 scan their row: online max/sum + top-3.
// ---------------------------------------------------------------------------
__global__ __launch_bounds__(32)
void k_out(const __bf16* __restrict__ A, const __bf16* __restrict__ BT,
           const float* __restrict__ bias,
           float* __restrict__ pmax, float* __restrict__ psum,
           float* __restrict__ ptv, int* __restrict__ pti) {
    const int tm    = blockIdx.x;   // 0..47
    const int split = blockIdx.y;   // 0..NSPLIT-1
    const int l     = threadIdx.x & 31;

    __shared__ __align__(16) __bf16 Alds[16 * HID];   // 32 KB activation panel
    __shared__ __align__(16) float  tile[16 * 64];    // 4 KB logit tile

    // ---- async stage of the A panel into LDS (16 rows x 1024 bf16) ----
    {
        unsigned long long gsrc =
            (unsigned long long)(const void*)(A + (long)tm * 16 * HID);
        unsigned lbase = (unsigned)(size_t)&Alds[0];
#pragma unroll 4
        for (int it = 0; it < 64; ++it) {          // 64 * 32 lanes * 16B = 32KB
            unsigned off = (unsigned)(it * 512 + l * 16);
            unsigned long long ga = gsrc + off;
            unsigned la = lbase + off;
            asm volatile("global_load_async_to_lds_b128 %0, %1, off"
                         :: "v"(la), "v"(ga) : "memory");
        }
        asm volatile("s_wait_asynccnt 0x0" ::: "memory");
        __syncthreads();
    }

    float rmax = -3.0e38f, rsum = 0.0f;
    float tv0 = -3.0e38f, tv1 = -3.0e38f, tv2 = -3.0e38f;
    int   ti0 = 0, ti1 = 0, ti2 = 0;

    const int ar = l & 15;          // fragment row (A) / col (B)
    const int kb = (l >> 4) * 8;    // fragment K base
    const int g0 = split * GROUPS_PER_SPLIT;

    for (int g = g0; g < g0 + GROUPS_PER_SPLIT; ++g) {   // 64 cols per group
        v8f acc[4] = {};
        const __bf16* Bb = BT + (long)g * 64 * HID;
        for (int k = 0; k < HID; k += 32) {
            v16bf a;
#pragma unroll
            for (int i = 0; i < 8; ++i) {
                const int kk = (i < 4) ? (kb + 2 * i) : (kb + 8 + 2 * i);
                a[2 * i]     = Alds[ar * HID + k + kk];
                a[2 * i + 1] = Alds[ar * HID + k + kk + 1];
            }
#pragma unroll
            for (int j = 0; j < 4; ++j) {
                v16bf b = ld_frag(Bb + (long)j * 16 * HID + k, HID);
                acc[j] = wmma_bf16(a, b, acc[j]);
            }
        }
        // ---- epilogue: 16x64 logits -> LDS, per-row online stats ----
        const int col = l & 15;
        const int rb  = (l >> 4) * 8;
#pragma unroll
        for (int j = 0; j < 4; ++j) {
            const float bv = bias[g * 64 + j * 16 + col];
#pragma unroll
            for (int e = 0; e < 8; ++e)
                tile[(rb + e) * 64 + j * 16 + col] = acc[j][e] + bv;
        }
        __syncthreads();
        if (l < 16) {
#pragma unroll 8
            for (int c = 0; c < 64; ++c) {
                float v  = tile[l * 64 + c];
                int  idx = g * 64 + c;
                float m2 = fmaxf(rmax, v);
                rsum = rsum * __expf(rmax - m2) + __expf(v - m2);
                rmax = m2;
                if (v > tv0)      { tv2=tv1; ti2=ti1; tv1=tv0; ti1=ti0; tv0=v; ti0=idx; }
                else if (v > tv1) { tv2=tv1; ti2=ti1; tv1=v;  ti1=idx; }
                else if (v > tv2) { tv2=v;  ti2=idx; }
            }
        }
        __syncthreads();
    }
    if (l < 16) {
        int row = tm * 16 + l;
        int o   = row * NSPLIT + split;
        pmax[o] = rmax;
        psum[o] = rsum;
        ptv[o * 3 + 0] = tv0; ptv[o * 3 + 1] = tv1; ptv[o * 3 + 2] = tv2;
        pti[o * 3 + 0] = ti0; pti[o * 3 + 1] = ti1; pti[o * 3 + 2] = ti2;
    }
}

// ---------------------------------------------------------------------------
// Per-batch beam update: merge split stats, build 9 candidates (finished
// beams frozen to (cum, EOS)), select top-3, gather sequences.
// ---------------------------------------------------------------------------
__global__ void k_merge(const float* __restrict__ pmax, const float* __restrict__ psum,
                        const float* __restrict__ ptv, const int* __restrict__ pti,
                        const float* __restrict__ cum, const int* __restrict__ fin,
                        const int* __restrict__ seqs_cur, int* __restrict__ seqs_next,
                        float* __restrict__ cum_next, int* __restrict__ fin_next,
                        int* __restrict__ tok_next, int* __restrict__ sel_beam,
                        int* __restrict__ sel_fin, int t) {
    int b = blockIdx.x;
    if (threadIdx.x != 0) return;

    float cv[9]; int cb[9], ct[9];
    for (int k = 0; k < KBEAM; ++k) {
        int row = b * KBEAM + k;
        if (fin[row]) {
            cv[3 * k] = cum[row]; cb[3 * k] = k; ct[3 * k] = EOS_T;
            cv[3 * k + 1] = -1.0f; cb[3 * k + 1] = k; ct[3 * k + 1] = EOS_T;
            cv[3 * k + 2] = -1.0f; cb[3 * k + 2] = k; ct[3 * k + 2] = EOS_T;
            continue;
        }
        float M = -3.0e38f;
        for (int s = 0; s < NSPLIT; ++s) M = fmaxf(M, pmax[row * NSPLIT + s]);
        float S = 0.0f;
        for (int s = 0; s < NSPLIT; ++s)
            S += psum[row * NSPLIT + s] * __expf(pmax[row * NSPLIT + s] - M);
        float t0 = -3.0e38f, t1 = -3.0e38f, t2 = -3.0e38f;
        int i0 = 0, i1 = 0, i2 = 0;
        for (int s = 0; s < NSPLIT; ++s) {
            for (int j = 0; j < 3; ++j) {
                float v = ptv[(row * NSPLIT + s) * 3 + j];
                int   i = pti[(row * NSPLIT + s) * 3 + j];
                if (v > t0)      { t2=t1; i2=i1; t1=t0; i1=i0; t0=v; i0=i; }
                else if (v > t1) { t2=t1; i2=i1; t1=v;  i1=i; }
                else if (v > t2) { t2=v;  i2=i; }
            }
        }
        float c = cum[row];
        float inv = (S > 0.0f) ? (1.0f / S) : 0.0f;
        cv[3 * k]     = __expf(t0 - M) * inv * c; cb[3 * k]     = k; ct[3 * k]     = i0;
        cv[3 * k + 1] = __expf(t1 - M) * inv * c; cb[3 * k + 1] = k; ct[3 * k + 1] = i1;
        cv[3 * k + 2] = __expf(t2 - M) * inv * c; cb[3 * k + 2] = k; ct[3 * k + 2] = i2;
    }
    float bv[3] = {-4.0e38f, -4.0e38f, -4.0e38f};
    int   bb[3] = {0, 0, 0}, bt[3] = {EOS_T, EOS_T, EOS_T};
    for (int i = 0; i < 9; ++i) {
        float v = cv[i];
        if (v > bv[0]) { bv[2]=bv[1]; bb[2]=bb[1]; bt[2]=bt[1];
                         bv[1]=bv[0]; bb[1]=bb[0]; bt[1]=bt[0];
                         bv[0]=v; bb[0]=cb[i]; bt[0]=ct[i]; }
        else if (v > bv[1]) { bv[2]=bv[1]; bb[2]=bb[1]; bt[2]=bt[1];
                              bv[1]=v; bb[1]=cb[i]; bt[1]=ct[i]; }
        else if (v > bv[2]) { bv[2]=v; bb[2]=cb[i]; bt[2]=ct[i]; }
    }
    for (int j = 0; j < KBEAM; ++j) {
        int r    = b * KBEAM + j;
        int src  = b * KBEAM + bb[j];
        int fsel = fin[src];
        sel_beam[r] = bb[j];
        sel_fin[r]  = fsel;
        tok_next[r] = bt[j];
        cum_next[r] = bv[j];
        fin_next[r] = (fsel || bt[j] == EOS_T) ? 1 : 0;
        for (int tt = 0; tt < TLEN; ++tt)
            seqs_next[r * TLEN + tt] = seqs_cur[src * TLEN + tt];
        seqs_next[r * TLEN + t] = bt[j];
    }
}

// ---------------------------------------------------------------------------
// Hidden-state beam gather: finished beams keep old hidden.
// ---------------------------------------------------------------------------
__global__ void k_sel_hidden(const float* __restrict__ hidden_cur,
                             const float* __restrict__ hnew,
                             const int* __restrict__ sel_beam,
                             const int* __restrict__ sel_fin,
                             float* __restrict__ hidden_next) {
    int r = blockIdx.x;
    int src = (r / KBEAM) * KBEAM + sel_beam[r];
    const float* s = sel_fin[r] ? (hidden_cur + (long)src * HID)
                                : (hnew + (long)src * HID);
    for (int i = threadIdx.x; i < HID; i += blockDim.x)
        hidden_next[(long)r * HID + i] = s[i];
}

// ---------------------------------------------------------------------------
// Final: best beam per batch -> tokens (int32 bits) + score.
// ---------------------------------------------------------------------------
__global__ void k_final(const float* __restrict__ cum, const int* __restrict__ seqs,
                        int* __restrict__ out_tok, float* __restrict__ out_sc) {
    int b = blockIdx.x;
    if (threadIdx.x != 0) return;
    int best = 0; float bv = cum[b * KBEAM];
    for (int k = 1; k < KBEAM; ++k)
        if (cum[b * KBEAM + k] > bv) { bv = cum[b * KBEAM + k]; best = k; }
    out_sc[b] = bv;
    for (int tt = 0; tt < TLEN; ++tt)
        out_tok[b * TLEN + tt] = seqs[(b * KBEAM + best) * TLEN + tt];
}

// ---------------------------------------------------------------------------
extern "C" void kernel_launch(void* const* d_in, const int* in_sizes, int n_in,
                              void* d_out, int out_size, void* d_ws, size_t ws_size,
                              hipStream_t stream) {
    const float* embed = (const float*)d_in[0];
    const float* Wx    = (const float*)d_in[1];
    const float* Wh    = (const float*)d_in[2];
    const float* bx    = (const float*)d_in[3];
    const float* bh    = (const float*)d_in[4];
    const float* Wout  = (const float*)d_in[5];
    const float* bout  = (const float*)d_in[6];
    const float* enc   = (const float*)d_in[7];
    (void)in_sizes; (void)n_in; (void)out_size; (void)ws_size;

    char* w = (char*)d_ws;
    auto alloc = [&](size_t bytes) -> void* {
        void* p = (void*)w;
        w += (bytes + 255) & ~(size_t)255;
        return p;
    };

    __bf16* WxT   = (__bf16*)alloc((size_t)G3H * XIN * 2);   // [3072][1536]
    __bf16* WhT   = (__bf16*)alloc((size_t)G3H * HID * 2);   // [3072][1024]
    __bf16* WoutT = (__bf16*)alloc((size_t)VOC * HID * 2);   // [32000][1024]
    __bf16* Ax    = (__bf16*)alloc((size_t)NROW * XIN * 2);
    __bf16* Hb    = (__bf16*)alloc((size_t)NROW * HID * 2);
    float*  hidA  = (float*)alloc((size_t)NROW * HID * 4);
    float*  hidB  = (float*)alloc((size_t)NROW * HID * 4);
    float*  gx    = (float*)alloc((size_t)NROW * G3H * 4);
    float*  gh    = (float*)alloc((size_t)NROW * G3H * 4);
    float*  hnew  = (float*)alloc((size_t)NROW * HID * 4);
    __bf16* hnewb = (__bf16*)alloc((size_t)NROW * HID * 2);
    float*  pmax  = (float*)alloc((size_t)NROW * NSPLIT * 4);
    float*  psum  = (float*)alloc((size_t)NROW * NSPLIT * 4);
    float*  ptv   = (float*)alloc((size_t)NROW * NSPLIT * 3 * 4);
    int*    pti   = (int*)alloc((size_t)NROW * NSPLIT * 3 * 4);
    int*    tokA  = (int*)alloc((size_t)NROW * 4);
    int*    tokB  = (int*)alloc((size_t)NROW * 4);
    float*  cumA  = (float*)alloc((size_t)NROW * 4);
    float*  cumB  = (float*)alloc((size_t)NROW * 4);
    int*    finA  = (int*)alloc((size_t)NROW * 4);
    int*    finB  = (int*)alloc((size_t)NROW * 4);
    int*    seqA  = (int*)alloc((size_t)NROW * TLEN * 4);
    int*    seqB  = (int*)alloc((size_t)NROW * TLEN * 4);
    int*    selB  = (int*)alloc((size_t)NROW * 4);
    int*    selF  = (int*)alloc((size_t)NROW * 4);

    // --- weight conversion + transpose (bf16) ---
    {
        long n = (long)G3H * XIN;
        k_cvtT<<<dim3((unsigned)((n + 255) / 256)), dim3(256), 0, stream>>>(Wx, WxT, XIN, G3H);
        n = (long)G3H * HID;
        k_cvtT<<<dim3((unsigned)((n + 255) / 256)), dim3(256), 0, stream>>>(Wh, WhT, HID, G3H);
        n = (long)VOC * HID;
        k_cvtT<<<dim3((unsigned)((n + 255) / 256)), dim3(256), 0, stream>>>(Wout, WoutT, HID, VOC);
    }

    // --- init state ---
    k_init<<<dim3(NROW), dim3(256), 0, stream>>>(enc, hidA, Ax, tokA, cumA, finA, seqA);

    int*   tok_cur = tokA,  *tok_nxt = tokB;
    float* cum_cur = cumA,  *cum_nxt = cumB;
    int*   fin_cur = finA,  *fin_nxt = finB;
    int*   seq_cur = seqA,  *seq_nxt = seqB;
    float* hid_cur = hidA,  *hid_nxt = hidB;

    for (int t = 0; t < TLEN; ++t) {
        // gather embeddings + hidden -> bf16 A matrices
        k_gather_x<<<dim3(NROW), dim3(256), 0, stream>>>(embed, hid_cur, tok_cur, Ax, Hb);

        // GRU gate GEMMs: gx = Ax @ Wx + bx ; gh = Hb @ Wh + bh  (64x64 tiles)
        k_gemm64<<<dim3(G3H / 64, NROW / 64), dim3(32), 0, stream>>>(Ax, WxT, bx, gx, G3H, XIN);
        k_gemm64<<<dim3(G3H / 64, NROW / 64), dim3(32), 0, stream>>>(Hb, WhT, bh, gh, G3H, HID);

        // GRU combine
        k_gru<<<dim3((NROW * HID + 255) / 256), dim3(256), 0, stream>>>(gx, gh, hid_cur, hnew, hnewb);

        // vocab projection + online softmax + per-row top-3 (split over V)
        k_out<<<dim3(NROW / 16, NSPLIT), dim3(32), 0, stream>>>(hnewb, WoutT, bout, pmax, psum, ptv, pti);

        // beam update
        k_merge<<<dim3(BATCH), dim3(32), 0, stream>>>(pmax, psum, ptv, pti,
                                                      cum_cur, fin_cur, seq_cur, seq_nxt,
                                                      cum_nxt, fin_nxt, tok_nxt, selB, selF, t);

        // hidden gather per selected beam
        k_sel_hidden<<<dim3(NROW), dim3(256), 0, stream>>>(hid_cur, hnew, selB, selF, hid_nxt);

        // ping-pong
        { int* tp = tok_cur; tok_cur = tok_nxt; tok_nxt = tp; }
        { float* tp = cum_cur; cum_cur = cum_nxt; cum_nxt = tp; }
        { int* tp = fin_cur; fin_cur = fin_nxt; fin_nxt = tp; }
        { int* tp = seq_cur; seq_cur = seq_nxt; seq_nxt = tp; }
        { float* tp = hid_cur; hid_cur = hid_nxt; hid_nxt = tp; }
    }

    // finalize: out = [tokens int32 (256*12)] ++ [scores f32 (256)]
    int*   out_tok = (int*)d_out;
    float* out_sc  = (float*)d_out + BATCH * TLEN;
    k_final<<<dim3(BATCH), dim3(32), 0, stream>>>(cum_cur, seq_cur, out_tok, out_sc);
}
